// AttentionModule_3272765080034
// MI455X (gfx1250) — matrix operand order repaired and checked by reading, not compiled, not run on hardware
//
#include <hip/hip_runtime.h>
#include <hip/hip_bf16.h>

typedef __attribute__((ext_vector_type(16))) _Float16 v16h;
typedef __attribute__((ext_vector_type(8)))  _Float16 v8h;
typedef __attribute__((ext_vector_type(8)))  float    v8f;

#define BATCH 512
#define RDIM  100
#define HDIM  1024
#define VDIM  2048
#define O1DIM 1024
#define EDIM  512

// ---------------------------------------------------------------------------
// Fragment helpers.
// A-fragment (16x32 f16, M x K): lane L holds row M = L&15; element e maps to
// K = (L>>4)*16 + e  -> one contiguous 16-element run (two 8-element loads).
// B-fragment (32x16 f16, K x N): lane L holds col N = L&15, K = (L>>4)*16 + e.
// With B stored N-major (col x K), a fragment is one contiguous 32B load.
// ---------------------------------------------------------------------------
__device__ __forceinline__ v16h make_a_frag_f32(const float* __restrict__ rowptr, int kc) {
  v8f lo = *(const v8f*)(rowptr + kc);
  v8f hi = *(const v8f*)(rowptr + kc + 8);
  v16h a;
#pragma unroll
  for (int i = 0; i < 8; ++i) { a[i] = (_Float16)lo[i]; a[i + 8] = (_Float16)hi[i]; }
  return a;
}

__device__ __forceinline__ v16h make_a_frag_f16(const _Float16* __restrict__ rowptr, int kc) {
  v8h lo = *(const v8h*)(rowptr + kc);
  v8h hi = *(const v8h*)(rowptr + kc + 8);
  v16h a;
#pragma unroll
  for (int i = 0; i < 8; ++i) { a[i] = lo[i]; a[i + 8] = hi[i]; }
  return a;
}

// ---------------------------------------------------------------------------
// One-time weight prep: f32 (K x N, row-major) -> f16 (N x K, row-major).
// ---------------------------------------------------------------------------
__global__ __launch_bounds__(256) void cvt_transpose_kernel(
    const float* __restrict__ src, _Float16* __restrict__ dst, int K, int N) {
  const long long idx = (long long)blockIdx.x * 256 + threadIdx.x;
  if (idx >= (long long)K * N) return;
  const int k = (int)(idx / N);
  const int n = (int)(idx % N);
  dst[(size_t)n * K + k] = (_Float16)src[idx];
}

// ---------------------------------------------------------------------------
// hb[b, o] = h @ W1h + b1   (M=512, N=1024, K=1024), f32 output to workspace.
// ---------------------------------------------------------------------------
__global__ __launch_bounds__(256) void gemm_hb_kernel(
    const float* __restrict__ h, const _Float16* __restrict__ W1h_t,
    const float* __restrict__ b1, float* __restrict__ hb) {
  const int tid = threadIdx.x, lane = tid & 31, wave = tid >> 5;
  const int m0 = blockIdx.x * 16;
  const int n0 = wave * 128;
  const int half = lane >> 4, nl = lane & 15;
  const float* arow = h + (size_t)(m0 + nl) * HDIM + half * 16;

  v8f acc[8];
#pragma unroll
  for (int t = 0; t < 8; ++t) acc[t] = (v8f)0.0f;

  for (int kc = 0; kc < HDIM; kc += 32) {
    v16h a = make_a_frag_f32(arow, kc);
#pragma unroll
    for (int t = 0; t < 8; ++t) {
      const int col = n0 + t * 16 + nl;
      v16h bf = *(const v16h*)(W1h_t + (size_t)col * HDIM + kc + half * 16);
      acc[t] = __builtin_amdgcn_wmma_f32_16x16x32_f16(false, a, false, bf,
                                                      (short)0, acc[t], false, false);
    }
  }
#pragma unroll
  for (int t = 0; t < 8; ++t) {
    const int col = n0 + t * 16 + nl;
    const float bv = b1[col];
#pragma unroll
    for (int j = 0; j < 8; ++j) {
      const int m = m0 + half * 8 + j;
      hb[(size_t)m * O1DIM + col] = acc[t][j] + bv;
    }
  }
}

// ---------------------------------------------------------------------------
// Fused main GEMM + epilogue:
//   scores[m] = sum_o relu( (v@W1v)[m,o] + hb[b(m),o] ) * W2[o] + b2
// M-space = B*R = 51200 rows, N = 1024, K = 2048. x never hits memory.
// WG covers 32 rows x 1024 cols; each wave holds 2 M-tiles x 8 N-tiles
// (128 f32 accumulator VGPRs). Every B fragment feeds TWO WMMAs.
// Cross-wave score reduction is a fixed-order LDS sum (bitwise deterministic).
// ---------------------------------------------------------------------------
__global__ __launch_bounds__(256) void gemm_scores_kernel(
    const float* __restrict__ v, const _Float16* __restrict__ W1v_t,
    const float* __restrict__ hb, const float* __restrict__ W2,
    const float* __restrict__ b2, float* __restrict__ scores) {
  __shared__ float sred[8][32];
  const int tid = threadIdx.x, lane = tid & 31, wave = tid >> 5;
  const int m0 = blockIdx.x * 32;
  const int n0 = wave * 128;
  const int half = lane >> 4, nl = lane & 15;
  const float* arow0 = v + (size_t)(m0 + nl) * VDIM + half * 16;
  const float* arow1 = arow0 + (size_t)16 * VDIM;

  v8f acc0[8], acc1[8];
#pragma unroll
  for (int t = 0; t < 8; ++t) { acc0[t] = (v8f)0.0f; acc1[t] = (v8f)0.0f; }

  for (int kc = 0; kc < VDIM; kc += 32) {
    // pull the A (v) stream ahead of use: gfx1250 global_prefetch path
    __builtin_prefetch(arow0 + kc + 64, 0, 1);
    __builtin_prefetch(arow1 + kc + 64, 0, 1);
    v16h a0 = make_a_frag_f32(arow0, kc);
    v16h a1 = make_a_frag_f32(arow1, kc);
#pragma unroll
    for (int t = 0; t < 8; ++t) {
      const int col = n0 + t * 16 + nl;
      v16h bf = *(const v16h*)(W1v_t + (size_t)col * VDIM + kc + half * 16);
      acc0[t] = __builtin_amdgcn_wmma_f32_16x16x32_f16(false, a0, false, bf,
                                                       (short)0, acc0[t], false, false);
      acc1[t] = __builtin_amdgcn_wmma_f32_16x16x32_f16(false, a1, false, bf,
                                                       (short)0, acc1[t], false, false);
    }
  }

  // epilogue: add hb, relu, dot with W2, reduce to per-row scalars
  float rowsum0[8], rowsum1[8];
#pragma unroll
  for (int j = 0; j < 8; ++j) { rowsum0[j] = 0.0f; rowsum1[j] = 0.0f; }
#pragma unroll
  for (int t = 0; t < 8; ++t) {
    const int col = n0 + t * 16 + nl;
    const float w2 = W2[col];
#pragma unroll
    for (int j = 0; j < 8; ++j) {
      const unsigned m0j = (unsigned)(m0 + half * 8 + j);
      float x0 = acc0[t][j] + hb[(size_t)(m0j / 100u) * O1DIM + col];
      rowsum0[j] = fmaf(fmaxf(x0, 0.0f), w2, rowsum0[j]);
      const unsigned m1j = m0j + 16u;
      float x1 = acc1[t][j] + hb[(size_t)(m1j / 100u) * O1DIM + col];
      rowsum1[j] = fmaf(fmaxf(x1, 0.0f), w2, rowsum1[j]);
    }
  }
  // reduce across the 16 lanes of each half-wave (same row, different cols)
#pragma unroll
  for (int j = 0; j < 8; ++j) {
    float s0 = rowsum0[j], s1 = rowsum1[j];
#pragma unroll
    for (int off = 1; off < 16; off <<= 1) {
      s0 += __shfl_xor(s0, off, 16);
      s1 += __shfl_xor(s1, off, 16);
    }
    rowsum0[j] = s0; rowsum1[j] = s1;
  }
  // per-wave partials to private LDS rows (no contention), then a fixed-order
  // cross-wave sum: bitwise deterministic across replays.
  if (nl == 0) {
#pragma unroll
    for (int j = 0; j < 8; ++j) {
      sred[wave][half * 8 + j]      = rowsum0[j];
      sred[wave][16 + half * 8 + j] = rowsum1[j];
    }
  }
  __syncthreads();
  if (tid < 32) {
    float s = 0.0f;
#pragma unroll
    for (int w = 0; w < 8; ++w) s += sred[w][tid];
    scores[m0 + tid] = s + b2[0];
  }
}

// ---------------------------------------------------------------------------
// Softmax / log-softmax over R=100 per batch row.
// ---------------------------------------------------------------------------
__global__ __launch_bounds__(128) void softmax_kernel(
    const float* __restrict__ scores, float* __restrict__ alpha,
    float* __restrict__ attlog) {
  __shared__ float sh[128];
  const int b = blockIdx.x, tid = threadIdx.x;
  const float s = (tid < RDIM) ? scores[b * RDIM + tid] : -3.4e38f;
  sh[tid] = s;
  __syncthreads();
  for (int o = 64; o > 0; o >>= 1) {
    if (tid < o) sh[tid] = fmaxf(sh[tid], sh[tid + o]);
    __syncthreads();
  }
  const float mx = sh[0];
  __syncthreads();
  const float e = (tid < RDIM) ? expf(s - mx) : 0.0f;
  sh[tid] = e;
  __syncthreads();
  for (int o = 64; o > 0; o >>= 1) {
    if (tid < o) sh[tid] += sh[tid + o];
    __syncthreads();
  }
  const float sum = sh[0];
  if (tid < RDIM) {
    alpha[b * RDIM + tid]  = e / sum;
    attlog[b * RDIM + tid] = (s - mx) - logf(sum);
  }
}

// ---------------------------------------------------------------------------
// v_att[b, :] = sum_r v[b,r,:] * alpha[b,r]; also emit f16 copy for decoder.
// ---------------------------------------------------------------------------
__global__ __launch_bounds__(256) void vatt_kernel(
    const float* __restrict__ v, const float* __restrict__ alpha,
    float* __restrict__ v_att, _Float16* __restrict__ vatt_h) {
  __shared__ float al[RDIM];
  const int b = blockIdx.x, tid = threadIdx.x;
  if (tid < RDIM) al[tid] = alpha[b * RDIM + tid];
  __syncthreads();
  float acc[8];
#pragma unroll
  for (int c = 0; c < 8; ++c) acc[c] = 0.0f;
  const float* vb = v + (size_t)b * RDIM * VDIM;
  for (int r = 0; r < RDIM; ++r) {
    const float a = al[r];
    const float* vr = vb + (size_t)r * VDIM;
#pragma unroll
    for (int c = 0; c < 8; ++c) acc[c] = fmaf(vr[c * 256 + tid], a, acc[c]);
  }
#pragma unroll
  for (int c = 0; c < 8; ++c) {
    const int col = c * 256 + tid;
    v_att[(size_t)b * VDIM + col]  = acc[c];
    vatt_h[(size_t)b * VDIM + col] = (_Float16)acc[c];
  }
}

// ---------------------------------------------------------------------------
// v_att_dec = relu(v_att @ Wrec + brec)  (M=512, N=512, K=2048), WMMA f16.
// ---------------------------------------------------------------------------
__global__ __launch_bounds__(256) void gemm_dec_kernel(
    const _Float16* __restrict__ vatt_h, const _Float16* __restrict__ Wrec_t,
    const float* __restrict__ brec, float* __restrict__ dec) {
  const int tid = threadIdx.x, lane = tid & 31, wave = tid >> 5;
  const int m0 = blockIdx.x * 16;
  const int n0 = wave * 64;  // 4 tiles per wave
  const int half = lane >> 4, nl = lane & 15;
  const _Float16* arow = vatt_h + (size_t)(m0 + nl) * VDIM + half * 16;

  v8f acc[4];
#pragma unroll
  for (int t = 0; t < 4; ++t) acc[t] = (v8f)0.0f;

  for (int kc = 0; kc < VDIM; kc += 32) {
    v16h a = make_a_frag_f16(arow, kc);
#pragma unroll
    for (int t = 0; t < 4; ++t) {
      const int col = n0 + t * 16 + nl;
      v16h bf = *(const v16h*)(Wrec_t + (size_t)col * VDIM + kc + half * 16);
      acc[t] = __builtin_amdgcn_wmma_f32_16x16x32_f16(false, a, false, bf,
                                                      (short)0, acc[t], false, false);
    }
  }
#pragma unroll
  for (int t = 0; t < 4; ++t) {
    const int col = n0 + t * 16 + nl;
    const float bv = brec[col];
#pragma unroll
    for (int j = 0; j < 8; ++j) {
      const int m = m0 + half * 8 + j;
      dec[(size_t)m * EDIM + col] = fmaxf(acc[t][j] + bv, 0.0f);
    }
  }
}

// ---------------------------------------------------------------------------
extern "C" void kernel_launch(void* const* d_in, const int* in_sizes, int n_in,
                              void* d_out, int out_size, void* d_ws, size_t ws_size,
                              hipStream_t stream) {
  const float* h    = (const float*)d_in[0];
  const float* v    = (const float*)d_in[1];
  const float* W1   = (const float*)d_in[2];
  const float* b1   = (const float*)d_in[3];
  const float* W2   = (const float*)d_in[4];
  const float* b2   = (const float*)d_in[5];
  const float* Wrec = (const float*)d_in[6];
  const float* brec = (const float*)d_in[7];

  // workspace layout (bytes)
  char* ws = (char*)d_ws;
  _Float16* W1v_t  = (_Float16*)(ws);              // 1024 x 2048 f16 : 4 MB
  _Float16* W1h_t  = (_Float16*)(ws + 4194304);    // 1024 x 1024 f16 : 2 MB
  _Float16* Wrec_t = (_Float16*)(ws + 6291456);    //  512 x 2048 f16 : 2 MB
  float*    hb     = (float*)   (ws + 8388608);    //  512 x 1024 f32 : 2 MB
  float*    scores = (float*)   (ws + 10485760);   //  512 x 100  f32
  _Float16* vatt_h = (_Float16*)(ws + 10690560);   //  512 x 2048 f16 : 2 MB

  // output layout: v_att | v_att_dec | alpha | att_log
  float* out    = (float*)d_out;
  float* v_att  = out;
  float* dec    = out + 1048576;
  float* alpha  = out + 1310720;
  float* attlog = out + 1361920;

  // 1) one-time weight transpose + f32->f16 conversion
  cvt_transpose_kernel<<<(VDIM * O1DIM) / 256, 256, 0, stream>>>(W1, W1v_t, VDIM, O1DIM);
  cvt_transpose_kernel<<<(HDIM * O1DIM) / 256, 256, 0, stream>>>(W1 + (size_t)VDIM * O1DIM,
                                                                 W1h_t, HDIM, O1DIM);
  cvt_transpose_kernel<<<(VDIM * EDIM) / 256, 256, 0, stream>>>(Wrec, Wrec_t, VDIM, EDIM);

  // 2) hb = h @ W1h + b1
  gemm_hb_kernel<<<BATCH / 16, 256, 0, stream>>>(h, W1h_t, b1, hb);

  // 3) fused big GEMM -> scores (x never materialized), 32-row M-blocks
  gemm_scores_kernel<<<(BATCH * RDIM) / 32, 256, 0, stream>>>(v, W1v_t, hb, W2, b2, scores);

  // 4) softmax / log-softmax over R
  softmax_kernel<<<BATCH, 128, 0, stream>>>(scores, alpha, attlog);

  // 5) alpha-weighted sum over R (second and last full read of v)
  vatt_kernel<<<BATCH, 256, 0, stream>>>(v, alpha, v_att, vatt_h);

  // 6) decoder GEMM with ReLU epilogue
  gemm_dec_kernel<<<BATCH / 16, 256, 0, stream>>>(vatt_h, Wrec_t, brec, dec);
}
